// DNATransportHamiltonianGNN_30477087933160
// MI455X (gfx1250) — matrix operand order repaired and checked by reading, not compiled, not run on hardware
//
#include <hip/hip_runtime.h>
#include <hip/hip_bf16.h>
#include <cmath>

// ---------------------------------------------------------------------------
// CDNA5 (gfx1250) wave32 WMMA types
// ---------------------------------------------------------------------------
typedef __attribute__((ext_vector_type(16))) __bf16 v16bf;
typedef __attribute__((ext_vector_type(8)))  float  v8f;

__device__ __forceinline__ unsigned short f2bf(float f) {
    union { float f; unsigned u; } x; x.f = f;
    unsigned u = x.u;
    u += 0x7FFFu + ((u >> 16) & 1u);   // round-to-nearest-even
    return (unsigned short)(u >> 16);
}

// ---------------------------------------------------------------------------
// Kernel 1: zero-fill the Hamiltonian (output-bandwidth dominant part)
// ---------------------------------------------------------------------------
__global__ void hm_zero_fill(float* __restrict__ p, size_t n) {
    size_t i      = (size_t)blockIdx.x * blockDim.x + threadIdx.x;
    size_t stride = (size_t)gridDim.x * blockDim.x;
    size_t n4 = n >> 2;
    float4* q = (float4*)p;
    const float4 z = make_float4(0.f, 0.f, 0.f, 0.f);
    for (size_t j = i; j < n4; j += stride) q[j] = z;
    // tail (H is even so normally empty)
    for (size_t j = (n4 << 2) + i; j < n; j += stride) p[j] = 0.f;
}

// ---------------------------------------------------------------------------
// Kernel 2: fused 128->128->ReLU->128->1 MLP via v_wmma_f32_16x16x32_bf16,
// then scatter into Hm.
//   mode 0: rows are DNA nodes (gathered as b*NPG + 2 + loc); write diagonal
//   mode 1: rows are edges; atomic-add symmetric off-diagonals
// ---------------------------------------------------------------------------
__global__ __launch_bounds__(256)
void mlp_wmma_scatter(const float* __restrict__ X, int M,
                      const float* __restrict__ W1, const float* __restrict__ b1,
                      const float* __restrict__ w2, const float* __restrict__ b2,
                      float* __restrict__ Hm,
                      int mode, int H, int NPG,
                      const int* __restrict__ edge_index, int E)
{
    __shared__ unsigned short W1t[128 * 128];  // bf16, [n][k] (transposed)
    __shared__ float b1s[128];
    __shared__ float w2s[128];

    const int tid = threadIdx.x;
    for (int i = tid; i < 128 * 128; i += blockDim.x) {
        int n = i >> 7, k = i & 127;
        W1t[i] = f2bf(W1[k * 128 + n]);        // W1 is [K][N] row-major
    }
    if (tid < 128) { b1s[tid] = b1[tid]; w2s[tid] = w2[tid]; }
    __syncthreads();
    const float b2v = b2[0];

    const int lane   = tid & 31;
    const int ncol   = lane & 15;              // column within 16x16 tile
    const int g      = lane >> 4;              // half-wave group
    const int wave   = (int)((blockIdx.x * blockDim.x + tid) >> 5);
    const int nwaves = (int)((gridDim.x * blockDim.x) >> 5);
    const int ntiles = (M + 15) >> 4;

    for (int t = wave; t < ntiles; t += nwaves) {
        const int  m     = t * 16 + ncol;      // A row handled by this lane
        const bool rowok = (m < M);

        size_t srow;
        if (mode == 0) { int b = m / H; srow = (size_t)b * NPG + 2 + (m - b * H); }
        else           { srow = (size_t)m; }
        const float* xr = X + srow * 128;

        // prefetch next tile's feature row (global_prefetch_b8)
        {
            int tn = t + nwaves;
            if (tn < ntiles) {
                int m2 = tn * 16 + ncol;
                if (m2 < M) {
                    size_t s2;
                    if (mode == 0) { int b = m2 / H; s2 = (size_t)b * NPG + 2 + (m2 - b * H); }
                    else           { s2 = (size_t)m2; }
                    __builtin_prefetch((const void*)(X + s2 * 128), 0, 0);
                }
            }
        }

        // --- A tiles (16x32 bf16, ISA layout): lane holds K in {8g..8g+7} U {16+8g..16+8g+7}
        union { v16bf v; unsigned short s[16]; } a[4];
        #pragma unroll
        for (int ks = 0; ks < 4; ++ks) {
            const int k0 = ks * 32 + 8 * g;
            if (rowok) {
                float4 c0 = *(const float4*)(xr + k0);
                float4 c1 = *(const float4*)(xr + k0 + 4);
                float4 c2 = *(const float4*)(xr + k0 + 16);
                float4 c3 = *(const float4*)(xr + k0 + 20);
                a[ks].s[0]  = f2bf(c0.x); a[ks].s[1]  = f2bf(c0.y);
                a[ks].s[2]  = f2bf(c0.z); a[ks].s[3]  = f2bf(c0.w);
                a[ks].s[4]  = f2bf(c1.x); a[ks].s[5]  = f2bf(c1.y);
                a[ks].s[6]  = f2bf(c1.z); a[ks].s[7]  = f2bf(c1.w);
                a[ks].s[8]  = f2bf(c2.x); a[ks].s[9]  = f2bf(c2.y);
                a[ks].s[10] = f2bf(c2.z); a[ks].s[11] = f2bf(c2.w);
                a[ks].s[12] = f2bf(c3.x); a[ks].s[13] = f2bf(c3.y);
                a[ks].s[14] = f2bf(c3.z); a[ks].s[15] = f2bf(c3.w);
            } else {
                #pragma unroll
                for (int j = 0; j < 16; ++j) a[ks].s[j] = 0;
            }
        }

        float racc[8];
        #pragma unroll
        for (int r = 0; r < 8; ++r) racc[r] = 0.f;

        // --- loop N tiles of layer-1 output; fold ReLU + layer-2 dot
        #pragma unroll
        for (int nt = 0; nt < 8; ++nt) {
            const int n = nt * 16 + ncol;
            union { v8f v; float f[8]; } c;
            const float bb = b1s[n];           // bias enters via C: C(m,n)=b1[n]
            #pragma unroll
            for (int r = 0; r < 8; ++r) c.f[r] = bb;

            #pragma unroll
            for (int ks = 0; ks < 4; ++ks) {
                // B tile (32x16 bf16): lane holds K in {8g..8g+7} U {16+8g..16+8g+7}
                union { v16bf v; uint4 q[2]; } bm;
                const int base = n * 128 + ks * 32 + g * 8;
                bm.q[0] = *(const uint4*)&W1t[base];
                bm.q[1] = *(const uint4*)&W1t[base + 16];
                c.v = __builtin_amdgcn_wmma_f32_16x16x32_bf16(
                        false, a[ks].v, false, bm.v, (short)0, c.v, false, false);
            }
            const float wn = w2s[n];
            #pragma unroll
            for (int r = 0; r < 8; ++r) {
                float y = c.f[r];
                y = y > 0.f ? y : 0.f;         // ReLU
                racc[r] += y * wn;             // layer-2 partial (this lane's column)
            }
        }

        // --- butterfly reduce over the 16 lanes of each half-wave
        #pragma unroll
        for (int off = 1; off < 16; off <<= 1) {
            #pragma unroll
            for (int r = 0; r < 8; ++r) racc[r] += __shfl_xor(racc[r], off, 32);
        }

        // --- scatter (D layout: half g holds rows 8g..8g+7)
        if (ncol == 0) {
            #pragma unroll
            for (int r = 0; r < 8; ++r) {
                const int mm = t * 16 + r + 8 * g;
                if (mm >= M) continue;
                const float val = racc[r] + b2v;
                if (mode == 0) {
                    int b = mm / H; int loc = mm - b * H;
                    size_t o = ((size_t)b * H + (size_t)loc) * H + loc;
                    Hm[o] = val + 1e-6f;       // onsite + epsilon*I
                } else {
                    int src = edge_index[mm];
                    int dst = edge_index[E + mm];
                    int b  = src / NPG;
                    int sl = src - b * NPG - 2;
                    int dl = dst - b * NPG - 2;
                    if (sl >= 0 && dl >= 0 && sl != dl) {   // dna_mask & u!=v
                        int u = sl < dl ? sl : dl;
                        int v = sl < dl ? dl : sl;
                        size_t base0 = (size_t)b * H * (size_t)H;
                        atomicAdd(Hm + base0 + (size_t)u * H + v, val);
                        atomicAdd(Hm + base0 + (size_t)v * H + u, val);
                    }
                }
            }
        }
    }
}

// ---------------------------------------------------------------------------
// Host launcher
// ---------------------------------------------------------------------------
extern "C" void kernel_launch(void* const* d_in, const int* in_sizes, int n_in,
                              void* d_out, int out_size, void* d_ws, size_t ws_size,
                              hipStream_t stream)
{
    const int HID = 128;
    const float* node_features = (const float*)d_in[0];
    const float* edge_features = (const float*)d_in[1];
    const float* Wo1 = (const float*)d_in[3];
    const float* bo1 = (const float*)d_in[4];
    const float* Wo2 = (const float*)d_in[5];
    const float* bo2 = (const float*)d_in[6];
    const float* Wc1 = (const float*)d_in[7];
    const float* bc1 = (const float*)d_in[8];
    const float* Wc2 = (const float*)d_in[9];
    const float* bc2 = (const float*)d_in[10];
    const int* edge_index = (const int*)d_in[11];
    float* Hm = (float*)d_out;

    const long N = (long)in_sizes[0] / HID;   // total nodes
    const int  E = in_sizes[1] / HID;         // edges
    const long out = (long)out_size;

    // derive (batch_size B, H=num_dna) from out == B*H*H and N == B*(H+2)
    int H = 0, Bsz = 0;
    for (int b = 1; b <= 4096; ++b) {
        if (out % b) continue;
        long hh = out / b;
        long h  = (long)(sqrt((double)hh) + 0.5);
        if (h * h != hh) continue;
        if ((long)b * (h + 2) == N) { Bsz = b; H = (int)h; break; }
    }
    if (!H) { H = 2048; Bsz = (int)(N / (H + 2)); }
    const int NPG = H + 2;

    // 1) zero the 134 MB output (bandwidth-bound stage)
    hm_zero_fill<<<4096, 256, 0, stream>>>(Hm, (size_t)out);

    // 2) onsite MLP -> diagonal
    const int Md = Bsz * H;
    int tiles0 = (Md + 15) / 16;
    int blk0 = (tiles0 + 7) / 8; if (blk0 > 1024) blk0 = 1024; if (blk0 < 1) blk0 = 1;
    mlp_wmma_scatter<<<blk0, 256, 0, stream>>>(node_features, Md,
                                               Wo1, bo1, Wo2, bo2,
                                               Hm, 0, H, NPG, nullptr, 0);

    // 3) coupling MLP -> symmetric off-diagonals
    int tiles1 = (E + 15) / 16;
    int blk1 = (tiles1 + 7) / 8; if (blk1 > 2048) blk1 = 2048; if (blk1 < 1) blk1 = 1;
    mlp_wmma_scatter<<<blk1, 256, 0, stream>>>(edge_features, E,
                                               Wc1, bc1, Wc2, bc2,
                                               Hm, 1, H, NPG, edge_index, E);
}
